// Mamba3Block_40235253629431
// MI455X (gfx1250) — compile-verified, matrix operand order checked by reading, not run on hardware
//
#include <hip/hip_runtime.h>
#include <hip/hip_bf16.h>
#include <math.h>

// ---------------------------------------------------------------------------
// Mamba-3 block for MI455X (gfx1250, wave32, WMMA).
// All heavy matmuls routed through v_wmma_f32_16x16x32_f16.
// ---------------------------------------------------------------------------

typedef __attribute__((ext_vector_type(16))) _Float16 v16h;
typedef __attribute__((ext_vector_type(8)))  float    v8f;
typedef _Float16 h16;

constexpr int DIM     = 1024;
constexpr int D_STATE = 32;
constexpr int HEADDIM = 64;
constexpr int D_INNER = 2048;
constexpr int NHEADS  = 32;
constexpr int CHUNK   = 64;
constexpr int NCHUNK  = 64;
constexpr int B_SZ    = 4;
constexpr int LSEQ    = 4096;
constexpr int DPROJ   = 2 * D_INNER + 2 * D_STATE + NHEADS; // 4192
constexpr int BL      = B_SZ * LSEQ;                        // 16384

__device__ __forceinline__ float siluf(float x) { return x / (1.f + __expf(-x)); }

__device__ __forceinline__ v8f wmma16(v16h a, v16h b, v8f c) {
    return __builtin_amdgcn_wmma_f32_16x16x32_f16(false, a, false, b, (short)0, c,
                                                  false, false);
}
__device__ __forceinline__ v8f v8f_zero() {
    v8f z = {0.f, 0.f, 0.f, 0.f, 0.f, 0.f, 0.f, 0.f};
    return z;
}

// ---------------------------------------------------------------------------
// 1) LayerNorm: x (f32) -> h (f16)
// ---------------------------------------------------------------------------
__global__ void k_layernorm(const float* __restrict__ x, const float* __restrict__ w,
                            const float* __restrict__ b, h16* __restrict__ h) {
    int bl = blockIdx.x;
    const float* xr = x + (size_t)bl * DIM;
    __shared__ float s1[256], s2[256];
    float a = 0.f, sq = 0.f;
    for (int i = threadIdx.x; i < DIM; i += 256) {
        float v = xr[i]; a += v; sq += v * v;
    }
    s1[threadIdx.x] = a; s2[threadIdx.x] = sq;
    __syncthreads();
    for (int off = 128; off; off >>= 1) {
        if (threadIdx.x < off) {
            s1[threadIdx.x] += s1[threadIdx.x + off];
            s2[threadIdx.x] += s2[threadIdx.x + off];
        }
        __syncthreads();
    }
    float mu  = s1[0] / (float)DIM;
    float var = s2[0] / (float)DIM - mu * mu;
    float inv = rsqrtf(var + 1e-6f);
    h16* hr = h + (size_t)bl * DIM;
    for (int i = threadIdx.x; i < DIM; i += 256)
        hr[i] = (h16)(((xr[i] - mu) * inv) * w[i] + b[i]);
}

// ---------------------------------------------------------------------------
// 2) Weight transpose f32 (R x C) -> f16 (C x R) so GEMM B-fragments are
//    contiguous along K per lane.
// ---------------------------------------------------------------------------
__global__ void k_transpose_w(const float* __restrict__ src, h16* __restrict__ dst,
                              int R, int C) {
    size_t idx = (size_t)blockIdx.x * 256 + threadIdx.x;
    if (idx >= (size_t)R * C) return;
    int n = (int)(idx / R);
    int k = (int)(idx % R);
    dst[idx] = (h16)src[(size_t)k * C + n];
}

// ---------------------------------------------------------------------------
// 3/9) WMMA GEMM: A (M x K f16, row major) * BT (N x K f16, row major)^T.
//     Each wave: one 16x64 output tile, A fragment reused across 4 N-tiles.
// ---------------------------------------------------------------------------
template <bool ADD_RES, bool OUT_F16>
__global__ void k_gemm_wmma(const h16* __restrict__ A, const h16* __restrict__ BT,
                            const float* __restrict__ res, h16* __restrict__ outH,
                            float* __restrict__ outF, int M, int N, int K) {
    int wave   = blockIdx.x * (blockDim.x >> 5) + (threadIdx.x >> 5);
    int tilesN = (N + 63) >> 6;
    int tm = wave / tilesN;
    int tn = wave % tilesN;
    if (tm >= (M >> 4)) return;  // uniform per wave

    int lane = threadIdx.x & 31;
    int rl   = lane & 15;            // row (A) / col (B,D) within tile
    int kb   = (lane >> 4) << 4;     // K sub-offset per half-wave

    const h16* Arow = A + (size_t)(tm * 16 + rl) * K + kb;
    const h16* Brow[4];
#pragma unroll
    for (int j = 0; j < 4; ++j) {
        int col = tn * 64 + j * 16 + rl;
        if (col >= N) col = N - 1;   // clamp (stores are guarded below)
        Brow[j] = BT + (size_t)col * K + kb;
    }

    v8f acc[4];
#pragma unroll
    for (int j = 0; j < 4; ++j) acc[j] = v8f_zero();

#pragma unroll 4
    for (int k0 = 0; k0 < K; k0 += 32) {
        v16h a = *(const v16h*)(Arow + k0);
        __builtin_prefetch((const void*)(Arow + k0 + 256), 0, 1);
#pragma unroll
        for (int j = 0; j < 4; ++j) {
            v16h b = *(const v16h*)(Brow[j] + k0);
            acc[j] = wmma16(a, b, acc[j]);
        }
    }

    int orow = tm * 16 + ((lane >> 4) << 3);
#pragma unroll
    for (int j = 0; j < 4; ++j) {
        int ocol = tn * 64 + j * 16 + rl;
        if (ocol < N) {
#pragma unroll
            for (int i = 0; i < 8; ++i) {
                size_t oi = (size_t)(orow + i) * N + ocol;
                if constexpr (OUT_F16) {
                    outH[oi] = (h16)acc[j][i];
                } else {
                    float v = acc[j][i];
                    if constexpr (ADD_RES) v += res[oi];
                    outF[oi] = v;
                }
            }
        }
    }
}

// ---------------------------------------------------------------------------
// 4) Activations: z passthrough, xs=silu, Xc=xs*dt, Y init = D*xs,
//    B/C silu+rope, dt softplus, a = dt * (-exp(A_log)).
// ---------------------------------------------------------------------------
__global__ void k_activations(const h16* __restrict__ zx, const float* __restrict__ dt_bias,
                              const float* __restrict__ A_log, const float* __restrict__ Dp,
                              h16* __restrict__ zF, h16* __restrict__ XcF,
                              float* __restrict__ Ytot, h16* __restrict__ BkF,
                              h16* __restrict__ CkF, float* __restrict__ aF) {
    int bl = blockIdx.x;
    int b  = bl / LSEQ;
    int l  = bl % LSEQ;
    const h16* row = zx + (size_t)bl * DPROJ;
    __shared__ float dtS[NHEADS];
    int tid = threadIdx.x;

    if (tid < NHEADS) {
        float raw = (float)row[2 * D_INNER + 2 * D_STATE + tid] + dt_bias[tid];
        float dtv = raw > 20.f ? raw : log1pf(__expf(raw));
        dtS[tid] = dtv;
        aF[((size_t)(b * NHEADS + tid)) * LSEQ + l] = dtv * (-__expf(A_log[tid]));
    }
    __syncthreads();

    for (int d = tid; d < D_INNER; d += 256) {
        float zraw = (float)row[d];
        zF[(size_t)bl * D_INNER + d] = (h16)zraw;
        float xs = siluf((float)row[D_INNER + d]);
        int hh = d >> 6;
        XcF[(size_t)bl * D_INNER + d]  = (h16)(xs * dtS[hh]);
        Ytot[(size_t)bl * D_INNER + d] = Dp[hh] * xs;   // D-skip term
    }

    if (tid < 32) {
        int isC = tid >> 4;       // 0 = B, 1 = C
        int u   = tid & 15;       // channel pair index
        const h16* seg = row + 2 * D_INNER + isC * D_STATE;
        float v1 = siluf((float)seg[2 * u]);
        float v2 = siluf((float)seg[2 * u + 1]);
        float o1, o2;
        if (u < 8) {  // ROPE_DIM = 16 -> 8 rotated pairs
            // inv_freq = 10000^{-(2u)/16};  ln(10000) = 9.210340372f
            float invf = __expf(-((float)(2 * u) / 16.f) * 9.210340372f);
            float ang  = (float)l * invf;
            float cs = __cosf(ang), sn = __sinf(ang);
            o1 = v1 * cs - v2 * sn;
            o2 = v1 * sn + v2 * cs;
        } else {
            o1 = v1; o2 = v2;
        }
        h16* dst = (isC ? CkF : BkF) + (size_t)bl * D_STATE;
        dst[2 * u]     = (h16)o1;
        dst[2 * u + 1] = (h16)o2;
    }
}

// ---------------------------------------------------------------------------
// 5) Intra-chunk SSD per (b, chunk, head). 4 waves / block.
//    scores = Ck * Bk^T (WMMA), M = scores * exp(segsum) masked (branchless),
//    Yd = M * Xc (WMMA), states = Bk^T * (decay*Xc) (WMMA).
// ---------------------------------------------------------------------------
__global__ __launch_bounds__(128) void k_chunk(
    const h16* __restrict__ BkF, const h16* __restrict__ CkF,
    const h16* __restrict__ XcF, const float* __restrict__ aF,
    float* __restrict__ Ytot, float* __restrict__ chunkStates,
    float* __restrict__ sumA) {
    int blk = blockIdx.x;
    int hh = blk & 31;
    int c  = (blk >> 5) & 63;
    int b  = blk >> 11;

    __shared__ __attribute__((aligned(32))) h16 CkS[64 * 32];
    __shared__ __attribute__((aligned(32))) h16 BkS[64 * 32];
    __shared__ __attribute__((aligned(32))) h16 BkTS[32 * 64];
    __shared__ __attribute__((aligned(32))) h16 XcTS[64 * 64];
    __shared__ __attribute__((aligned(32))) h16 dXcTS[64 * 64];
    __shared__ __attribute__((aligned(32))) h16 MS[64 * 64];
    __shared__ float acsS[64], decayS[64];

    int tid = threadIdx.x;
    size_t bcBase = (size_t)(b * LSEQ + c * CHUNK);

    for (int idx = tid; idx < 64 * 32; idx += 128) {
        int l = idx >> 5, n = idx & 31;
        h16 bv = BkF[(bcBase + l) * D_STATE + n];
        BkS[idx]          = bv;
        BkTS[n * 64 + l]  = bv;
        CkS[idx]          = CkF[(bcBase + l) * D_STATE + n];
    }
    for (int idx = tid; idx < 64 * 64; idx += 128) {
        int l = idx >> 6, p = idx & 63;
        XcTS[p * 64 + l] = XcF[(bcBase + l) * D_INNER + hh * HEADDIM + p];
    }
    if (tid == 0) {
        float cs = 0.f;
        const float* ap = aF + ((size_t)(b * NHEADS + hh)) * LSEQ + c * CHUNK;
        for (int l = 0; l < 64; ++l) { cs += ap[l]; acsS[l] = cs; }
        sumA[((size_t)(b * NHEADS + hh)) * NCHUNK + c] = cs;
    }
    __syncthreads();
    if (tid < 64) decayS[tid] = __expf(acsS[63] - acsS[tid]);
    __syncthreads();
    for (int idx = tid; idx < 64 * 64; idx += 128) {
        int l = idx & 63;
        dXcTS[idx] = (h16)((float)XcTS[idx] * decayS[l]);
    }
    __syncthreads();

    int w = tid >> 5, lane = tid & 31;
    int rl = lane & 15;
    int kb = (lane >> 4) << 4;
    int lbase = w * 16 + ((lane >> 4) << 3);   // D-fragment row base

    // scores -> masked, decayed M (each wave: rows 16w..16w+15); branchless mask
    {
        v16h aC = *(const v16h*)&CkS[(w * 16 + rl) * 32 + kb];
        float lacs[8];
#pragma unroll
        for (int i = 0; i < 8; ++i) lacs[i] = acsS[lbase + i];
#pragma unroll
        for (int st = 0; st < 4; ++st) {
            v16h bB = *(const v16h*)&BkS[(st * 16 + rl) * 32 + kb];
            v8f acc = v8f_zero();
            acc = wmma16(aC, bB, acc);
            int s = st * 16 + rl;
            float acs_s = acsS[s];
#pragma unroll
            for (int i = 0; i < 8; ++i) {
                int l = lbase + i;
                float m = acc[i] * __expf(lacs[i] - acs_s);
                m = (s <= l) ? m : 0.f;           // v_cndmask, no branch
                MS[l * 64 + s] = (h16)m;
            }
        }
    }
    __syncthreads();

    // Yd = M * Xc  -> accumulate into Ytot
#pragma unroll
    for (int pt = 0; pt < 4; ++pt) {
        v8f acc = v8f_zero();
#pragma unroll
        for (int k0 = 0; k0 < 64; k0 += 32) {
            v16h aM = *(const v16h*)&MS[(w * 16 + rl) * 64 + k0 + kb];
            v16h bX = *(const v16h*)&XcTS[(pt * 16 + rl) * 64 + k0 + kb];
            acc = wmma16(aM, bX, acc);
        }
        int p = pt * 16 + rl;
        float* yrow = Ytot + (bcBase + lbase) * D_INNER + hh * HEADDIM + p;
#pragma unroll
        for (int i = 0; i < 8; ++i) yrow[(size_t)i * D_INNER] += acc[i];
    }

    // chunk states[n,p] = sum_l Bk[l,n]*decay[l]*Xc[l,p]
    size_t stBase = ((size_t)((b * NCHUNK + c) * NHEADS + hh)) * HEADDIM * D_STATE;
#pragma unroll
    for (int u = 0; u < 2; ++u) {
        int t  = w * 2 + u;
        int nt = t >> 2, pt = t & 3;
        v8f acc = v8f_zero();
#pragma unroll
        for (int k0 = 0; k0 < 64; k0 += 32) {
            v16h aB = *(const v16h*)&BkTS[(nt * 16 + rl) * 64 + k0 + kb];
            v16h bX = *(const v16h*)&dXcTS[(pt * 16 + rl) * 64 + k0 + kb];
            acc = wmma16(aB, bX, acc);
        }
        int nbase = nt * 16 + ((lane >> 4) << 3);
        int p = pt * 16 + rl;
        float* srow = chunkStates + stBase + (size_t)p * D_STATE + nbase;
#pragma unroll
        for (int i = 0; i < 8; ++i) srow[i] = acc[i];
    }
}

// ---------------------------------------------------------------------------
// 6) Inter-chunk sequential state scan per (b, head).
// ---------------------------------------------------------------------------
__global__ void k_state_scan(const float* __restrict__ chunkStates,
                             const float* __restrict__ sumA, h16* __restrict__ stIn) {
    int bh = blockIdx.x;
    int b = bh >> 5, hh = bh & 31;
    int e = threadIdx.x * 8;
    float cur[8];
#pragma unroll
    for (int r = 0; r < 8; ++r) cur[r] = 0.f;
    for (int c = 0; c < NCHUNK; ++c) {
        size_t base = ((size_t)((b * NCHUNK + c) * NHEADS + hh)) * HEADDIM * D_STATE + e;
#pragma unroll
        for (int r = 0; r < 8; ++r) stIn[base + r] = (h16)cur[r];
        float dec = __expf(sumA[(size_t)bh * NCHUNK + c]);
#pragma unroll
        for (int r = 0; r < 8; ++r) cur[r] = cur[r] * dec + chunkStates[base + r];
    }
}

// ---------------------------------------------------------------------------
// 7) Yo = exp(Acs[l]) * (Ck * state_in^T), accumulated into Ytot. WMMA.
// ---------------------------------------------------------------------------
__global__ __launch_bounds__(128) void k_yo(const h16* __restrict__ CkF,
                                            const h16* __restrict__ stIn,
                                            const float* __restrict__ aF,
                                            float* __restrict__ Ytot) {
    int blk = blockIdx.x;
    int hh = blk & 31;
    int c  = (blk >> 5) & 63;
    int b  = blk >> 11;

    __shared__ __attribute__((aligned(32))) h16 CkS[64 * 32];
    __shared__ __attribute__((aligned(32))) h16 stS[HEADDIM * D_STATE];
    __shared__ float eacsS[64];

    int tid = threadIdx.x;
    size_t bcBase = (size_t)(b * LSEQ + c * CHUNK);

    for (int idx = tid; idx < 64 * 32; idx += 128)
        CkS[idx] = CkF[(bcBase + (idx >> 5)) * D_STATE + (idx & 31)];
    size_t stBase = ((size_t)((b * NCHUNK + c) * NHEADS + hh)) * HEADDIM * D_STATE;
    for (int idx = tid; idx < HEADDIM * D_STATE; idx += 128) stS[idx] = stIn[stBase + idx];
    if (tid == 0) {
        float cs = 0.f;
        const float* ap = aF + ((size_t)(b * NHEADS + hh)) * LSEQ + c * CHUNK;
        for (int l = 0; l < 64; ++l) { cs += ap[l]; eacsS[l] = __expf(cs); }
    }
    __syncthreads();

    int w = tid >> 5, lane = tid & 31;
    int rl = lane & 15;
    int kb = (lane >> 4) << 4;
    int lbase = w * 16 + ((lane >> 4) << 3);

    float eacs[8];
#pragma unroll
    for (int i = 0; i < 8; ++i) eacs[i] = eacsS[lbase + i];

    v16h aC = *(const v16h*)&CkS[(w * 16 + rl) * 32 + kb];
#pragma unroll
    for (int pt = 0; pt < 4; ++pt) {
        v16h bS = *(const v16h*)&stS[(pt * 16 + rl) * 32 + kb];
        v8f acc = v8f_zero();
        acc = wmma16(aC, bS, acc);
        int p = pt * 16 + rl;
        float* yrow = Ytot + (bcBase + lbase) * D_INNER + hh * HEADDIM + p;
#pragma unroll
        for (int i = 0; i < 8; ++i) yrow[(size_t)i * D_INNER] += eacs[i] * acc[i];
    }
}

// ---------------------------------------------------------------------------
// 8) Gate with silu(z) + RMSNorm -> yn (f16).
// ---------------------------------------------------------------------------
__global__ void k_gate_rmsnorm(const float* __restrict__ Ytot, const h16* __restrict__ zF,
                               const float* __restrict__ rms_w, h16* __restrict__ yn) {
    int bl = blockIdx.x;
    int tid = threadIdx.x;
    __shared__ float red[256];
    float vloc[8];
    float sq = 0.f;
#pragma unroll
    for (int r = 0; r < 8; ++r) {
        int d = tid + r * 256;
        float v = Ytot[(size_t)bl * D_INNER + d] * siluf((float)zF[(size_t)bl * D_INNER + d]);
        vloc[r] = v;
        sq += v * v;
    }
    red[tid] = sq;
    __syncthreads();
    for (int off = 128; off; off >>= 1) {
        if (tid < off) red[tid] += red[tid + off];
        __syncthreads();
    }
    float scale = rsqrtf(red[0] / (float)D_INNER + 1e-6f);
#pragma unroll
    for (int r = 0; r < 8; ++r) {
        int d = tid + r * 256;
        yn[(size_t)bl * D_INNER + d] = (h16)(vloc[r] * scale * rms_w[d]);
    }
}

// ---------------------------------------------------------------------------
// Launch
// ---------------------------------------------------------------------------
extern "C" void kernel_launch(void* const* d_in, const int* in_sizes, int n_in,
                              void* d_out, int out_size, void* d_ws, size_t ws_size,
                              hipStream_t stream) {
    const float* x       = (const float*)d_in[0];
    const float* ln_w    = (const float*)d_in[1];
    const float* ln_b    = (const float*)d_in[2];
    const float* W_in    = (const float*)d_in[3];
    const float* dt_bias = (const float*)d_in[4];
    const float* A_log   = (const float*)d_in[5];
    const float* Dp      = (const float*)d_in[6];
    const float* rms_w   = (const float*)d_in[7];
    const float* W_out   = (const float*)d_in[8];
    float* out = (float*)d_out;

    char* base = (char*)d_ws;
    size_t off = 0;
    auto alloc = [&](size_t bytes) {
        size_t o = off;
        off = (off + bytes + 255) & ~(size_t)255;
        return o;
    };

    h16*   hF16   = (h16*)(base + alloc((size_t)BL * DIM * 2));
    h16*   WinT   = (h16*)(base + alloc((size_t)DPROJ * DIM * 2));
    h16*   WoutT  = (h16*)(base + alloc((size_t)DIM * D_INNER * 2));
    h16*   zxF16  = (h16*)(base + alloc((size_t)BL * DPROJ * 2));
    h16*   zF16   = (h16*)(base + alloc((size_t)BL * D_INNER * 2));
    h16*   XcF16  = (h16*)(base + alloc((size_t)BL * D_INNER * 2));
    h16*   BkF16  = (h16*)(base + alloc((size_t)BL * D_STATE * 2));
    h16*   CkF16  = (h16*)(base + alloc((size_t)BL * D_STATE * 2));
    float* aF32   = (float*)(base + alloc((size_t)B_SZ * NHEADS * LSEQ * 4));
    float* sumA   = (float*)(base + alloc((size_t)B_SZ * NHEADS * NCHUNK * 4));
    float* Ytot   = (float*)(base + alloc((size_t)BL * D_INNER * 4));
    float* cStates= (float*)(base + alloc((size_t)B_SZ * NCHUNK * NHEADS * HEADDIM * D_STATE * 4));
    h16*   stIn   = (h16*)(base + alloc((size_t)B_SZ * NCHUNK * NHEADS * HEADDIM * D_STATE * 2));
    h16*   ynF16  = (h16*)(base + alloc((size_t)BL * D_INNER * 2));
    (void)ws_size; (void)in_sizes; (void)n_in; (void)out_size;

    // 1) LayerNorm
    k_layernorm<<<BL, 256, 0, stream>>>(x, ln_w, ln_b, hF16);

    // 2) Weight transposes (f32 -> f16, N x K layout)
    {
        size_t n1 = (size_t)DIM * DPROJ;
        k_transpose_w<<<(int)((n1 + 255) / 256), 256, 0, stream>>>(W_in, WinT, DIM, DPROJ);
        size_t n2 = (size_t)D_INNER * DIM;
        k_transpose_w<<<(int)((n2 + 255) / 256), 256, 0, stream>>>(W_out, WoutT, D_INNER, DIM);
    }

    // 3) GEMM1: h (BLxDIM) @ W_in (DIMxDPROJ) -> zxbcdt f16
    {
        int tilesN = (DPROJ + 63) / 64;              // 66
        int tiles  = (BL / 16) * tilesN;             // 67584
        int blocks = (tiles + 7) / 8;
        k_gemm_wmma<false, true><<<blocks, 256, 0, stream>>>(
            hF16, WinT, nullptr, zxF16, nullptr, BL, DPROJ, DIM);
    }

    // 4) Activations
    k_activations<<<BL, 256, 0, stream>>>(zxF16, dt_bias, A_log, Dp, zF16, XcF16,
                                          Ytot, BkF16, CkF16, aF32);

    // 5) Intra-chunk SSD (scores, Yd, chunk states)
    k_chunk<<<B_SZ * NCHUNK * NHEADS, 128, 0, stream>>>(BkF16, CkF16, XcF16, aF32,
                                                        Ytot, cStates, sumA);

    // 6) Inter-chunk state scan
    k_state_scan<<<B_SZ * NHEADS, 256, 0, stream>>>(cStates, sumA, stIn);

    // 7) Inter-chunk output Yo
    k_yo<<<B_SZ * NCHUNK * NHEADS, 128, 0, stream>>>(CkF16, stIn, aF32, Ytot);

    // 8) Gate + RMSNorm
    k_gate_rmsnorm<<<BL, 256, 0, stream>>>(Ytot, zF16, rms_w, ynF16);

    // 9) GEMM2: yn (BLxD_INNER) @ W_out (D_INNERxDIM) + residual x -> out f32
    {
        int tilesN = DIM / 64;                       // 16
        int tiles  = (BL / 16) * tilesN;             // 16384
        int blocks = (tiles + 7) / 8;
        k_gemm_wmma<true, false><<<blocks, 256, 0, stream>>>(
            ynF16, WoutT, x, nullptr, out, BL, DIM, D_INNER);
    }
}
